// MLPBaseline_81922206204130
// MI455X (gfx1250) — compile-verified, matrix operand order checked by reading, not run on hardware
//
#include <hip/hip_runtime.h>
#include <hip/hip_bf16.h>

// ---------------------------------------------------------------------------
// Problem constants (from reference)
// ---------------------------------------------------------------------------
#define NUM_TYPES 200
#define DIM_TOK   5001
#define SMALL_D   2
#define IN_DIM    5203          // NUM_TYPES + DIM_TOK + SMALL
#define K_REAL    (3 * IN_DIM)  // 15609
#define K_PAD     15616         // multiple of 32 for WMMA K-loop
#define HID       256
#define NB        512           // number of graphs

typedef _Float16 v8h  __attribute__((ext_vector_type(8)));
typedef _Float16 v16h __attribute__((ext_vector_type(16)));
typedef float    v8f  __attribute__((ext_vector_type(8)));

// ---------------------------------------------------------------------------
// Order-preserving float <-> uint encoding for atomicMax on shared memory
// ---------------------------------------------------------------------------
__device__ __forceinline__ unsigned fenc(float f) {
    unsigned u = __float_as_uint(f);
    return (u & 0x80000000u) ? ~u : (u | 0x80000000u);
}
__device__ __forceinline__ float fdec(unsigned e) {
    unsigned u = (e & 0x80000000u) ? (e ^ 0x80000000u) : ~e;
    return __uint_as_float(u);
}

__device__ __forceinline__ int lower_bound_dev(const int* a, int n, int v) {
    int lo = 0, hi = n;
    while (lo < hi) { int mid = (lo + hi) >> 1; if (a[mid] < v) lo = mid + 1; else hi = mid; }
    return lo;
}

// ---------------------------------------------------------------------------
// K0: transpose + f32 -> f16 convert.  src [Kreal, N] row-major f32,
//     dst [N, Kpad] row-major f16 (zero padded rows k >= Kreal).
// ---------------------------------------------------------------------------
__global__ void transpose_to_f16(const float* __restrict__ src, _Float16* __restrict__ dst,
                                 int Kreal, int N, int Kpad) {
    long idx = (long)blockIdx.x * blockDim.x + threadIdx.x;
    long total = (long)N * Kpad;
    if (idx >= total) return;
    int n = (int)(idx / Kpad);
    int k = (int)(idx % Kpad);
    float v = (k < Kreal) ? src[(long)k * N + n] : 0.0f;
    dst[idx] = (_Float16)v;
}

// ---------------------------------------------------------------------------
// K1: per-segment pooled features -> H [NB, K_PAD] f16.
//     One workgroup per segment. batch[] is sorted -> contiguous range found
//     by binary search. LDS histograms for the one-hot dims.
// ---------------------------------------------------------------------------
__global__ void build_features(const int* __restrict__ x_type,
                               const int* __restrict__ x_tok,
                               const float* __restrict__ x_small,
                               const int* __restrict__ batch,
                               _Float16* __restrict__ H, int N) {
    __shared__ int      s_ct[NUM_TYPES];
    __shared__ int      s_ck[DIM_TOK];
    __shared__ float    s_sum[SMALL_D];
    __shared__ float    s_sq[SMALL_D];
    __shared__ unsigned s_mx[SMALL_D];
    __shared__ int      s_start, s_end;

    const int seg = blockIdx.x;
    const int tid = threadIdx.x;
    const int nth = blockDim.x;

    for (int i = tid; i < NUM_TYPES; i += nth) s_ct[i] = 0;
    for (int i = tid; i < DIM_TOK;   i += nth) s_ck[i] = 0;
    if (tid < SMALL_D) {
        s_sum[tid] = 0.0f; s_sq[tid] = 0.0f;
        s_mx[tid]  = fenc(-__builtin_inff());   // identity of max = -inf (matches jax)
    }
    if (tid == 0) {
        s_start = lower_bound_dev(batch, N, seg);
        s_end   = lower_bound_dev(batch, N, seg + 1);
    }
    __syncthreads();

    const int i0 = s_start, i1 = s_end;
    for (int i = i0 + tid; i < i1; i += nth) {
        int t = x_type[i];
        if (t >= 0 && t < NUM_TYPES) atomicAdd(&s_ct[t], 1);
        int k = x_tok[i];
        k = k < 0 ? 0 : (k > DIM_TOK - 1 ? DIM_TOK - 1 : k);   // jnp.clip
        atomicAdd(&s_ck[k], 1);
#pragma unroll
        for (int j = 0; j < SMALL_D; ++j) {
            float v = x_small[(long)i * SMALL_D + j];
            atomicAdd(&s_sum[j], v);
            atomicAdd(&s_sq[j], v * v);
            atomicMax(&s_mx[j], fenc(v));
        }
    }
    __syncthreads();

    const int   cnt = i1 - i0;
    const float inv = 1.0f / (float)(cnt > 0 ? cnt : 1);

    _Float16* hrow = H + (long)seg * K_PAD;
    for (int d = tid; d < K_PAD; d += nth) {
        float val = 0.0f;
        if (d < K_REAL) {
            int region = d / IN_DIM;      // 0=mean, 1=max, 2=std
            int idx    = d - region * IN_DIM;
            float mean, mx, msq;
            if (idx < NUM_TYPES) {
                int c = s_ct[idx];
                mean = (float)c * inv; mx = c > 0 ? 1.0f : 0.0f; msq = mean;
            } else if (idx < NUM_TYPES + DIM_TOK) {
                int c = s_ck[idx - NUM_TYPES];
                mean = (float)c * inv; mx = c > 0 ? 1.0f : 0.0f; msq = mean;
            } else {
                int j = idx - (NUM_TYPES + DIM_TOK);
                mean = s_sum[j] * inv;
                mx   = fdec(s_mx[j]);
                msq  = s_sq[j] * inv;
            }
            if (region == 0) val = mean;
            else if (region == 1) val = mx;
            else {
                float var = msq - mean * mean;
                var = var > 0.0f ? var : 0.0f;
                val = sqrtf(var + 1e-8f);
            }
        }
        hrow[d] = (_Float16)val;
    }
}

// ---------------------------------------------------------------------------
// GEMM tile kernel: one wave (32 threads) per 16x16 output tile.
//   A  : [M, Kpad] row-major f16
//   Bt : [Nn, Kpad] row-major f16  (i.e. B transposed; column n is a row)
//   C  = leaky_relu(A@B + bias)   -> f16 (OUT_HALF) or f32
// Fragment layout per CDNA5 ISA 7.12.2 (wave32):
//   A: lane(0..15)=row, half=lane>>4 ; VGPR0-3 = K[h*8 .. h*8+7],
//      VGPR4-7 = K[h*8+16 .. h*8+23]  -> two contiguous 16B loads.
//   B: lane(0..15)=col, same K split  -> contiguous since Bt is [N,K].
//   C/D: lane&15 = col, VGPR g = row g + 8*(lane>>4).
// ---------------------------------------------------------------------------
template <bool OUT_HALF>
__global__ void gemm_wmma_f16(const _Float16* __restrict__ A,
                              const _Float16* __restrict__ Bt,
                              const float* __restrict__ bias,
                              void* __restrict__ Cout,
                              int Kpad, int Nn) {
    const int n0   = blockIdx.x * 16;
    const int m0   = blockIdx.y * 16;
    const int lane = threadIdx.x & 31;
    const int r    = lane & 15;
    const int half = lane >> 4;

    const _Float16* pa = A  + (long)(m0 + r) * Kpad + half * 8;
    const _Float16* pb = Bt + (long)(n0 + r) * Kpad + half * 8;

    v8f c = {};
    for (int k = 0; k < Kpad; k += 32) {
        v8h alo = *(const v8h*)(pa + k);
        v8h ahi = *(const v8h*)(pa + k + 16);
        v8h blo = *(const v8h*)(pb + k);
        v8h bhi = *(const v8h*)(pb + k + 16);
        v16h a = __builtin_shufflevector(alo, ahi, 0,1,2,3,4,5,6,7,8,9,10,11,12,13,14,15);
        v16h b = __builtin_shufflevector(blo, bhi, 0,1,2,3,4,5,6,7,8,9,10,11,12,13,14,15);
        c = __builtin_amdgcn_wmma_f32_16x16x32_f16(
                /*neg_a=*/false, a, /*neg_b=*/false, b,
                /*c_mod=*/(short)0, c, /*reuse_a=*/false, /*reuse_b=*/false);
    }

    const int   col = n0 + r;
    const float bv  = bias[col];
#pragma unroll
    for (int g = 0; g < 8; ++g) {
        int   row = m0 + g + half * 8;
        float v   = c[g] + bv;
        v = v > 0.0f ? v : 0.01f * v;            // leaky_relu(0.01)
        if (OUT_HALF) ((_Float16*)Cout)[(long)row * Nn + col] = (_Float16)v;
        else          ((float*)Cout)[(long)row * Nn + col]    = v;
    }
}

// ---------------------------------------------------------------------------
// K4: out[b] = dot(H3[b, :], W3) + b3
// ---------------------------------------------------------------------------
__global__ void final_dot(const float* __restrict__ H3, const float* __restrict__ W3,
                          const float* __restrict__ b3, float* __restrict__ out) {
    __shared__ float s[HID];
    const int b = blockIdx.x;
    const int t = threadIdx.x;
    s[t] = H3[(long)b * HID + t] * W3[t];
    __syncthreads();
    for (int st = HID / 2; st > 0; st >>= 1) {
        if (t < st) s[t] += s[t + st];
        __syncthreads();
    }
    if (t == 0) out[b] = s[0] + b3[0];
}

// ---------------------------------------------------------------------------
// Host launcher
// ---------------------------------------------------------------------------
extern "C" void kernel_launch(void* const* d_in, const int* in_sizes, int n_in,
                              void* d_out, int out_size, void* d_ws, size_t ws_size,
                              hipStream_t stream) {
    const int*   x_type  = (const int*)d_in[0];
    const int*   x_tok   = (const int*)d_in[1];
    const float* x_small = (const float*)d_in[2];
    const int*   batch   = (const int*)d_in[3];
    const float* W1      = (const float*)d_in[4];
    const float* b1      = (const float*)d_in[5];
    const float* W2      = (const float*)d_in[6];
    const float* b2      = (const float*)d_in[7];
    const float* W3      = (const float*)d_in[8];
    const float* b3      = (const float*)d_in[9];
    float*       out     = (float*)d_out;

    const int N = in_sizes[0];

    // Workspace carve-up (all offsets 256B aligned).
    char* ws = (char*)d_ws;
    const size_t SZ_W1T = (size_t)HID * K_PAD * sizeof(_Float16);  // 7,995,392
    const size_t SZ_H   = (size_t)NB  * K_PAD * sizeof(_Float16);  // 15,990,784
    const size_t SZ_W2T = (size_t)HID * HID   * sizeof(_Float16);  //   131,072
    const size_t SZ_H2  = (size_t)NB  * HID   * sizeof(_Float16);  //   262,144

    _Float16* W1T = (_Float16*)(ws);
    _Float16* Hf  = (_Float16*)(ws + SZ_W1T);
    _Float16* W2T = (_Float16*)(ws + SZ_W1T + SZ_H);
    _Float16* H2  = (_Float16*)(ws + SZ_W1T + SZ_H + SZ_W2T);
    float*    H3  = (float*)   (ws + SZ_W1T + SZ_H + SZ_W2T + SZ_H2);

    // K0a: W1 [K_REAL, 256] f32 -> W1T [256, K_PAD] f16
    {
        long total = (long)HID * K_PAD;
        int  grid  = (int)((total + 255) / 256);
        transpose_to_f16<<<grid, 256, 0, stream>>>(W1, W1T, K_REAL, HID, K_PAD);
    }
    // K0b: W2 [256, 256] f32 -> W2T [256, 256] f16
    {
        long total = (long)HID * HID;
        int  grid  = (int)((total + 255) / 256);
        transpose_to_f16<<<grid, 256, 0, stream>>>(W2, W2T, HID, HID, HID);
    }
    // K1: segment pooling -> H [NB, K_PAD] f16
    build_features<<<NB, 256, 0, stream>>>(x_type, x_tok, x_small, batch, Hf, N);

    // K2: H [512, K_PAD] @ W1 -> leaky -> H2 [512, 256] f16
    {
        dim3 grid(HID / 16, NB / 16);   // (16, 32) wave-tiles
        gemm_wmma_f16<true><<<grid, 32, 0, stream>>>(Hf, W1T, b1, (void*)H2, K_PAD, HID);
    }
    // K3: H2 [512, 256] @ W2 -> leaky -> H3 [512, 256] f32
    {
        dim3 grid(HID / 16, NB / 16);
        gemm_wmma_f16<false><<<grid, 32, 0, stream>>>(H2, W2T, b2, (void*)H3, HID, HID);
    }
    // K4: out[b] = H3[b,:] . W3 + b3
    final_dot<<<NB, HID, 0, stream>>>(H3, W3, b3, out);

    (void)n_in; (void)out_size; (void)ws_size;
}
